// TGN_55748675502602
// MI455X (gfx1250) — compile-verified, matrix operand order checked by reading, not run on hardware
//
#include <hip/hip_runtime.h>
#include <hip/hip_bf16.h>

typedef __attribute__((ext_vector_type(16))) __bf16 v16bf;
typedef __attribute__((ext_vector_type(8)))  __bf16 v8bf;
typedef __attribute__((ext_vector_type(8)))  float  v8f;
typedef __attribute__((ext_vector_type(4)))  float  f32x4;

// ---- problem constants (from reference) ----
constexpr int Dc  = 128, Tc = 128, Ec = 100;
constexpr int Bc  = 2048;
constexpr int N0c = 3 * Bc;          // 6144
constexpr int K0c = 5;
constexpr int N1c = N0c * K0c;       // 30720
constexpr int K1c = 10;
constexpr int NNc = 200000, Mc = 4096, MSGc = 2 * Dc + Ec + Tc; // 484

// K index inside a 32-wide chunk for the bf16 WMMA A/B lane layout (ISA 7.12.2):
// lanes 0-15: K = {0..7, 16..23}; lanes 16-31: K = {8..15, 24..31}
__device__ __forceinline__ int kmap(int lane, int j) {
  return ((j < 8) ? j : (j + 8)) + ((lane & 16) ? 8 : 0);
}

// ---------------------------------------------------------------------------
// Per-wave GEMM. A assembled on the fly: af8(row, g0, dst8) fills 8 consecutive
// K elements starting at g0 (g0 always a multiple of 8). B pre-packed in WMMA
// lane layout. B fragments are batched 8 tiles at a time so WMMAs issue
// back-to-back after a single load wait.
// ---------------------------------------------------------------------------
template <int CHUNKS, int NTILES, typename AF8>
__device__ __forceinline__ void wave_gemm(AF8 af8, const __bf16* __restrict__ B,
                                          float* o, const float* __restrict__ bias,
                                          int lane) {
  static_assert(NTILES % 8 == 0 || NTILES == 8, "tile grouping");
  v8f acc[NTILES];
#pragma unroll
  for (int t = 0; t < NTILES; ++t)
#pragma unroll
    for (int j = 0; j < 8; ++j) acc[t][j] = 0.f;
  const int row = lane & 15;
  const int hi8 = (lane & 16) ? 8 : 0;
#pragma unroll 1
  for (int c = 0; c < CHUNKS; ++c) {
    float lo[8], hh[8];
    const int g0 = c * 32 + hi8;
    af8(row, g0, lo);
    af8(row, g0 + 16, hh);
    v16bf a;
#pragma unroll
    for (int j = 0; j < 8; ++j) { a[j] = (__bf16)lo[j]; a[8 + j] = (__bf16)hh[j]; }
#pragma unroll
    for (int t0 = 0; t0 < NTILES; t0 += 8) {
      v16bf b[8];
#pragma unroll
      for (int u = 0; u < 8; ++u)
        b[u] = *(const v16bf*)(B + (((size_t)c * NTILES + t0 + u) * 32 + lane) * 16);
#pragma unroll
      for (int u = 0; u < 8; ++u)
        acc[t0 + u] = __builtin_amdgcn_wmma_f32_16x16x32_bf16(
            false, a, false, b[u], (short)0, acc[t0 + u], false, false);
    }
  }
  const int rbase = (lane & 16) ? 8 : 0;
  const int col0  = lane & 15;
#pragma unroll
  for (int t = 0; t < NTILES; ++t)
#pragma unroll
    for (int j = 0; j < 8; ++j) {
      int col = t * 16 + col0;
      o[(j + rbase) * (NTILES * 16) + col] = acc[t][j] + bias[col];
    }
}

// Dual-B version (K and V share the same A assembly / Time2Vec cos work)
template <int CHUNKS, int NTILES, typename AF8>
__device__ __forceinline__ void wave_gemm_dual(AF8 af8,
    const __bf16* __restrict__ B0, const __bf16* __restrict__ B1,
    float* o0, float* o1,
    const float* __restrict__ bias0, const float* __restrict__ bias1, int lane) {
  v8f acc0[NTILES], acc1[NTILES];
#pragma unroll
  for (int t = 0; t < NTILES; ++t)
#pragma unroll
    for (int j = 0; j < 8; ++j) { acc0[t][j] = 0.f; acc1[t][j] = 0.f; }
  const int row = lane & 15;
  const int hi8 = (lane & 16) ? 8 : 0;
#pragma unroll 1
  for (int c = 0; c < CHUNKS; ++c) {
    float lo[8], hh[8];
    const int g0 = c * 32 + hi8;
    af8(row, g0, lo);
    af8(row, g0 + 16, hh);
    v16bf a;
#pragma unroll
    for (int j = 0; j < 8; ++j) { a[j] = (__bf16)lo[j]; a[8 + j] = (__bf16)hh[j]; }
#pragma unroll
    for (int t0 = 0; t0 < NTILES; t0 += 8) {
      v16bf b[8];
#pragma unroll
      for (int u = 0; u < 8; ++u)
        b[u] = *(const v16bf*)(B0 + (((size_t)c * NTILES + t0 + u) * 32 + lane) * 16);
#pragma unroll
      for (int u = 0; u < 8; ++u)
        acc0[t0 + u] = __builtin_amdgcn_wmma_f32_16x16x32_bf16(
            false, a, false, b[u], (short)0, acc0[t0 + u], false, false);
#pragma unroll
      for (int u = 0; u < 8; ++u)
        b[u] = *(const v16bf*)(B1 + (((size_t)c * NTILES + t0 + u) * 32 + lane) * 16);
#pragma unroll
      for (int u = 0; u < 8; ++u)
        acc1[t0 + u] = __builtin_amdgcn_wmma_f32_16x16x32_bf16(
            false, a, false, b[u], (short)0, acc1[t0 + u], false, false);
    }
  }
  const int rbase = (lane & 16) ? 8 : 0;
  const int col0  = lane & 15;
#pragma unroll
  for (int t = 0; t < NTILES; ++t)
#pragma unroll
    for (int j = 0; j < 8; ++j) {
      int col = t * 16 + col0, r = j + rbase;
      o0[r * (NTILES * 16) + col] = acc0[t][j] + bias0[col];
      o1[r * (NTILES * 16) + col] = acc1[t][j] + bias1[col];
    }
}

// ---------------------------------------------------------------------------
// Pack a weight matrix (fp32) into bf16 WMMA B-operand layout.
// transpose==0: w(k,n) = W[k*ld + n]   (W is [Kdim, N])
// transpose==1: w(k,n) = W[n*ld + k]   (W is [N, Kdim], e.g. GRU W_ih used as x@W^T)
// ---------------------------------------------------------------------------
__global__ void pack_b_kernel(const float* __restrict__ W, int Kdim, int Kpad, int N,
                              int transpose, int ld, __bf16* __restrict__ out) {
  int idx = blockIdx.x * blockDim.x + threadIdx.x;
  int total = Kpad * N;
  if (idx >= total) return;
  int j    = idx & 15;
  int lane = (idx >> 4) & 31;
  int rest = idx >> 9;
  int ntiles = N >> 4;
  int t = rest % ntiles;
  int c = rest / ntiles;
  int k = c * 32 + kmap(lane, j);
  int n = t * 16 + (lane & 15);
  float v = 0.f;
  if (k < Kdim) v = transpose ? W[(size_t)n * ld + k] : W[(size_t)k * ld + n];
  out[idx] = (__bf16)v;
}

__global__ void init_map_kernel(int* __restrict__ map, int n) {
  int i = blockIdx.x * blockDim.x + threadIdx.x;
  if (i < n) map[i] = -1;
}
__global__ void scatter_map_kernel(const int* __restrict__ nids, int* __restrict__ map, int m) {
  int i = blockIdx.x * blockDim.x + threadIdx.x;
  if (i < m) map[nids[i]] = i;
}

// ---------------------------------------------------------------------------
// GRU memory update for the M updated rows. One wave per 16 messages.
// ---------------------------------------------------------------------------
__global__ void __launch_bounds__(32) gru_kernel(
    const float* __restrict__ msgs, const int* __restrict__ msg_nids,
    const float* __restrict__ memory,
    const __bf16* __restrict__ BWih, const __bf16* __restrict__ BWhh,
    const float* __restrict__ bih, const float* __restrict__ bhh,
    float* __restrict__ upd) {
  __shared__ float gi[16 * 384];
  __shared__ float gh[16 * 384];
  __shared__ float hrow[16 * 128];
  const int lane = threadIdx.x;
  const int mt   = blockIdx.x;
  for (int e = lane; e < 16 * 128; e += 32) {
    int r = e >> 7, c = e & 127;
    hrow[e] = memory[(size_t)msg_nids[mt * 16 + r] * Dc + c];
  }
  __syncthreads();
  auto fx8 = [&](int r, int g0, float* dst) {
    const float* rowp = msgs + (size_t)(mt * 16 + r) * MSGc;
    if (g0 + 8 <= MSGc) {
      f32x4 x0 = *(const f32x4*)(rowp + g0), x1 = *(const f32x4*)(rowp + g0 + 4);
#pragma unroll
      for (int j = 0; j < 4; ++j) { dst[j] = x0[j]; dst[4 + j] = x1[j]; }
    } else {
#pragma unroll
      for (int e = 0; e < 8; ++e) { int k = g0 + e; dst[e] = (k < MSGc) ? rowp[k] : 0.f; }
    }
  };
  wave_gemm<16, 24>(fx8, BWih, gi, bih, lane);         // K=484 -> 512
  auto fh8 = [&](int r, int g0, float* dst) {
    f32x4 x0 = *(const f32x4*)(hrow + r * 128 + g0);
    f32x4 x1 = *(const f32x4*)(hrow + r * 128 + g0 + 4);
#pragma unroll
    for (int j = 0; j < 4; ++j) { dst[j] = x0[j]; dst[4 + j] = x1[j]; }
  };
  wave_gemm<4, 24>(fh8, BWhh, gh, bhh, lane);          // K=128
  __syncthreads();
  for (int e = lane; e < 16 * 128; e += 32) {
    int r = e >> 7, c = e & 127;
    float ir = gi[r * 384 + c], iz = gi[r * 384 + 128 + c], in = gi[r * 384 + 256 + c];
    float hr = gh[r * 384 + c], hz = gh[r * 384 + 128 + c], hn = gh[r * 384 + 256 + c];
    float rg = 1.f / (1.f + __expf(-(ir + hr)));
    float u  = 1.f / (1.f + __expf(-(iz + hz)));
    float nn = tanhf(in + rg * hn);
    float h  = hrow[e];
    upd[(size_t)(mt * 16 + r) * Dc + c] = (1.f - u) * nn + u * h;
  }
}

// ---------------------------------------------------------------------------
// Q projection: q = [mem[seed] | cos(time_b)] @ Wq + bq (also emits node feats bf16)
// ---------------------------------------------------------------------------
__global__ void __launch_bounds__(32) qproj_kernel(
    const float* __restrict__ memory, const float* __restrict__ upd,
    const int* __restrict__ map, const int* __restrict__ seed,
    const float* __restrict__ time_b,
    const __bf16* __restrict__ BWq, const float* __restrict__ bq,
    float* __restrict__ qout, __bf16* __restrict__ nf) {
  __shared__ float oL[16 * 128];
  const int lane = threadIdx.x;
  const int mt   = blockIdx.x;
  auto af8 = [&](int r, int g0, float* dst) {
    int node = mt * 16 + r;
    if (g0 < Dc) {                          // node feature run (128 is 8-aligned)
      int s  = seed[node];
      int mi = map[s];
      const float* src = (mi >= 0) ? upd + (size_t)mi * Dc : memory + (size_t)s * Dc;
      f32x4 x0 = *(const f32x4*)(src + g0), x1 = *(const f32x4*)(src + g0 + 4);
#pragma unroll
      for (int j = 0; j < 4; ++j) { dst[j] = x0[j]; dst[4 + j] = x1[j]; }
#pragma unroll
      for (int j = 0; j < 8; ++j) nf[(size_t)node * Dc + g0 + j] = (__bf16)dst[j];
    } else {                                // Time2Vec at t=0 -> cos(time_b)
      int j0 = g0 - Dc;
      f32x4 b0 = *(const f32x4*)(time_b + j0), b1 = *(const f32x4*)(time_b + j0 + 4);
#pragma unroll
      for (int j = 0; j < 4; ++j) { dst[j] = cosf(b0[j]); dst[4 + j] = cosf(b1[j]); }
    }
  };
  wave_gemm<8, 8>(af8, BWq, oL, bq, lane);
  __syncthreads();
  for (int e = lane; e < 16 * 128; e += 32)
    qout[(size_t)(mt * 16 + (e >> 7)) * Dc + (e & 127)] = oL[e];
}

// ---------------------------------------------------------------------------
// Fused attention core: one wave per node. Builds kin tile (neighbors padded to
// 16 rows, K padded 356->384), K/V WMMA GEMMs, masked softmax, weighted V.
// ---------------------------------------------------------------------------
template <int KN>
__global__ void __launch_bounds__(64) attn_kernel(
    const float* __restrict__ qv, const float* __restrict__ nbrsrc,
    const int* __restrict__ nbr_local,
    const float* __restrict__ edge, const float* __restrict__ times,
    const float* __restrict__ nbr_times,
    const float* __restrict__ time_w, const float* __restrict__ time_b,
    const int* __restrict__ mask,
    const __bf16* __restrict__ BWk, const __bf16* __restrict__ BWv,
    const float* __restrict__ bk, const float* __restrict__ bv,
    __bf16* __restrict__ hout) {
  __shared__ float kl[2][16 * 128];
  __shared__ float vl[2][16 * 128];
  __shared__ float sl[2][2][16];
  const int wid  = threadIdx.x >> 5;
  const int lane = threadIdx.x & 31;
  const int node = blockIdx.x * 2 + wid;
  // scalar fallback (straddling runs 224..231, 352..359 and K padding 360..383)
  auto af1 = [&](int r, int g) -> float {
    if (g < Dc + Ec) return edge[((size_t)node * KN + r) * Ec + (g - Dc)];
    if (g < Dc + Ec + Tc) {
      int j = g - (Dc + Ec);
      float dt = times[node] - nbr_times[(size_t)node * KN + r];
      return cosf(dt * time_w[j] + time_b[j]);
    }
    return 0.f;
  };
  auto af8 = [&](int r, int g0, float* dst) {
    if (r >= KN) {
#pragma unroll
      for (int j = 0; j < 8; ++j) dst[j] = 0.f;
      return;
    }
    if (g0 + 8 <= Dc) {                          // neighbor node features
      if (!nbrsrc) {
#pragma unroll
        for (int j = 0; j < 8; ++j) dst[j] = 0.f;  // hop-1: zeros
      } else {
        int nl = nbr_local[(size_t)node * KN + r] - N0c;
        const float* src = nbrsrc + (size_t)nl * Dc + g0;
        f32x4 x0 = *(const f32x4*)src, x1 = *(const f32x4*)(src + 4);
#pragma unroll
        for (int j = 0; j < 4; ++j) { dst[j] = x0[j]; dst[4 + j] = x1[j]; }
      }
    } else if (g0 >= Dc && g0 + 8 <= Dc + Ec) {  // edge features, 16B aligned
      const float* src = edge + ((size_t)node * KN + r) * Ec + (g0 - Dc);
      f32x4 x0 = *(const f32x4*)src, x1 = *(const f32x4*)(src + 4);
#pragma unroll
      for (int j = 0; j < 4; ++j) { dst[j] = x0[j]; dst[4 + j] = x1[j]; }
    } else if (g0 >= Dc + Ec && g0 + 8 <= Dc + Ec + Tc) {  // Time2Vec
      int j0 = g0 - (Dc + Ec);
      float dt = times[node] - nbr_times[(size_t)node * KN + r];
      f32x4 w0 = *(const f32x4*)(time_w + j0), w1 = *(const f32x4*)(time_w + j0 + 4);
      f32x4 b0 = *(const f32x4*)(time_b + j0), b1 = *(const f32x4*)(time_b + j0 + 4);
#pragma unroll
      for (int j = 0; j < 4; ++j) {
        dst[j]     = cosf(dt * w0[j] + b0[j]);
        dst[4 + j] = cosf(dt * w1[j] + b1[j]);
      }
    } else {                                     // straddling runs / padding
#pragma unroll
      for (int e = 0; e < 8; ++e) dst[e] = af1(r, g0 + e);
    }
  };
  wave_gemm_dual<12, 8>(af8, BWk, BWv, kl[wid], vl[wid], bk, bv, lane);
  __syncthreads();
  {
    int h = lane >> 4, r = lane & 15;
    float s = -1e9f;
    if (r < KN && mask[(size_t)node * KN + r] > 0) {
      float acc = 0.f;
      const float* qrow = qv + (size_t)node * Dc + h * 64;
      const float* krow = &kl[wid][r * 128 + h * 64];
      for (int d = 0; d < 64; ++d) acc += qrow[d] * krow[d];
      s = acc * 0.125f;                          // 1/sqrt(64)
    }
    sl[wid][h][r] = s;
  }
  __syncthreads();
#pragma unroll
  for (int i = 0; i < 4; ++i) {
    int d = lane + 32 * i;
    int h = d >> 6;
    float mx = -1e30f;
    for (int r = 0; r < 16; ++r) mx = fmaxf(mx, sl[wid][h][r]);
    float sum = 0.f, acc = 0.f;
    for (int r = 0; r < 16; ++r) {
      float e = __expf(sl[wid][h][r] - mx);
      sum += e;
      acc += e * vl[wid][r * 128 + d];
    }
    hout[(size_t)node * Dc + d] = (__bf16)(acc / sum);
  }
}

// ---------------------------------------------------------------------------
// Output projection: out = relu([hout | node_feat] @ Wo + bo)  (fp32 out)
// ---------------------------------------------------------------------------
__global__ void __launch_bounds__(32) oproj_kernel(
    const __bf16* __restrict__ hout, const __bf16* __restrict__ nf,
    const __bf16* __restrict__ BWo, const float* __restrict__ bo,
    float* __restrict__ out) {
  __shared__ float oL[16 * 128];
  const int lane = threadIdx.x;
  const int mt   = blockIdx.x;
  auto af8 = [&](int r, int g0, float* dst) {
    int node = mt * 16 + r;
    const __bf16* src = (g0 < Dc) ? (hout + (size_t)node * Dc + g0)
                                  : (nf + (size_t)node * Dc + (g0 - Dc));
    v8bf x = *(const v8bf*)src;
#pragma unroll
    for (int j = 0; j < 8; ++j) dst[j] = (float)x[j];
  };
  wave_gemm<8, 8>(af8, BWo, oL, bo, lane);
  __syncthreads();
  for (int e = lane; e < 16 * 128; e += 32)
    out[(size_t)(mt * 16 + (e >> 7)) * Dc + (e & 127)] = fmaxf(oL[e], 0.f);
}

// ---------------------------------------------------------------------------
// Link predictor: sigmoid(relu(a@Ws + bs + c@Wd + bd) @ Wo + bo)
// ---------------------------------------------------------------------------
__global__ void __launch_bounds__(128) linkpred_kernel(
    const float* __restrict__ z0, const int* __restrict__ src,
    const int* __restrict__ cand,
    const float* __restrict__ Ws, const float* __restrict__ bs,
    const float* __restrict__ Wd, const float* __restrict__ bd,
    const float* __restrict__ Wo, const float* __restrict__ bo,
    float* __restrict__ outp) {
  __shared__ float red[128];
  const int d = threadIdx.x;
  const int pair = blockIdx.x;
  const float* a = z0 + (size_t)src[pair] * Dc;
  const float* c = z0 + (size_t)cand[pair] * Dc;
  float acc = bs[d] + bd[d];
  for (int i = 0; i < Dc; ++i) acc += a[i] * Ws[i * Dc + d] + c[i] * Wd[i * Dc + d];
  red[d] = fmaxf(acc, 0.f) * Wo[d];
  __syncthreads();
  for (int s = 64; s > 0; s >>= 1) {
    if (d < s) red[d] += red[d + s];
    __syncthreads();
  }
  if (d == 0) outp[pair] = 1.f / (1.f + __expf(-(red[0] + bo[0])));
}

// ---------------------------------------------------------------------------
extern "C" void kernel_launch(void* const* d_in, const int* in_sizes, int n_in,
                              void* d_out, int out_size, void* d_ws, size_t ws_size,
                              hipStream_t stream) {
  const float* memory   = (const float*)d_in[0];
  const float* time_w   = (const float*)d_in[1];
  const float* time_b   = (const float*)d_in[2];
  const float* gru_Wih  = (const float*)d_in[3];
  const float* gru_bih  = (const float*)d_in[4];
  const float* gru_Whh  = (const float*)d_in[5];
  const float* gru_bhh  = (const float*)d_in[6];
  const float* Wq0 = (const float*)d_in[7],  *bq0 = (const float*)d_in[8];
  const float* Wk0 = (const float*)d_in[9],  *bk0 = (const float*)d_in[10];
  const float* Wv0 = (const float*)d_in[11], *bv0 = (const float*)d_in[12];
  const float* Wo0 = (const float*)d_in[13], *bo0 = (const float*)d_in[14];
  const float* Wq1 = (const float*)d_in[15], *bq1 = (const float*)d_in[16];
  const float* Wk1 = (const float*)d_in[17], *bk1 = (const float*)d_in[18];
  const float* Wv1 = (const float*)d_in[19], *bv1 = (const float*)d_in[20];
  const float* Wo1 = (const float*)d_in[21], *bo1 = (const float*)d_in[22];
  const float* lp_Ws = (const float*)d_in[23], *lp_bs = (const float*)d_in[24];
  const float* lp_Wd = (const float*)d_in[25], *lp_bd = (const float*)d_in[26];
  const float* lp_Wo = (const float*)d_in[27], *lp_bo = (const float*)d_in[28];
  const float* msgs       = (const float*)d_in[29];
  const float* times0     = (const float*)d_in[30];
  const float* nbr_times0 = (const float*)d_in[31];
  const float* nbr_feats0 = (const float*)d_in[32];
  const float* times1     = (const float*)d_in[33];
  const float* nbr_times1 = (const float*)d_in[34];
  const float* nbr_feats1 = (const float*)d_in[35];
  const int* msg_nids   = (const int*)d_in[36];
  const int* seed0      = (const int*)d_in[37];
  const int* seed1      = (const int*)d_in[38];
  const int* nbr_local0 = (const int*)d_in[39];
  const int* nbr_mask0  = (const int*)d_in[40];
  const int* nbr_mask1  = (const int*)d_in[41];
  const int* src_local  = (const int*)d_in[42];
  const int* dst_local  = (const int*)d_in[43];
  const int* neg_local  = (const int*)d_in[44];

  // ---- workspace carve-up ----
  char* p = (char*)d_ws;
  auto alloc = [&](size_t bytes) -> void* {
    void* r = (void*)p;
    p += (bytes + 255) & ~(size_t)255;
    return r;
  };
  int*    map   = (int*)alloc((size_t)NNc * 4);
  float*  upd   = (float*)alloc((size_t)Mc * Dc * 4);
  __bf16* BWih  = (__bf16*)alloc((size_t)512 * 384 * 2);
  __bf16* BWhh  = (__bf16*)alloc((size_t)128 * 384 * 2);
  __bf16* BWq[2], *BWk[2], *BWv[2], *BWo[2];
  for (int l = 0; l < 2; ++l) {
    BWq[l] = (__bf16*)alloc((size_t)256 * 128 * 2);
    BWk[l] = (__bf16*)alloc((size_t)384 * 128 * 2);
    BWv[l] = (__bf16*)alloc((size_t)384 * 128 * 2);
    BWo[l] = (__bf16*)alloc((size_t)256 * 128 * 2);
  }
  float*  qbuf  = (float*)alloc((size_t)N1c * Dc * 4);
  __bf16* nf    = (__bf16*)alloc((size_t)N1c * Dc * 2);
  __bf16* hout  = (__bf16*)alloc((size_t)N1c * Dc * 2);
  float*  out1  = (float*)alloc((size_t)N1c * Dc * 4);
  float*  out0  = (float*)alloc((size_t)N0c * Dc * 4);

  // ---- updated-memory index map ----
  init_map_kernel<<<(NNc + 255) / 256, 256, 0, stream>>>(map, NNc);
  scatter_map_kernel<<<(Mc + 255) / 256, 256, 0, stream>>>(msg_nids, map, Mc);

  // ---- weight packing into WMMA B layout ----
  auto pack = [&](const float* W, int Kdim, int Kpad, int N, int tr, int ld, __bf16* out) {
    int total = Kpad * N;
    pack_b_kernel<<<(total + 255) / 256, 256, 0, stream>>>(W, Kdim, Kpad, N, tr, ld, out);
  };
  pack(gru_Wih, MSGc, 512, 384, 1, MSGc, BWih);   // x @ Wih^T
  pack(gru_Whh, Dc,   128, 384, 1, Dc,   BWhh);   // h @ Whh^T
  const float* Wq_[2] = {Wq0, Wq1}; const float* Wk_[2] = {Wk0, Wk1};
  const float* Wv_[2] = {Wv0, Wv1}; const float* Wo_[2] = {Wo0, Wo1};
  for (int l = 0; l < 2; ++l) {
    pack(Wq_[l], 256, 256, 128, 0, 128, BWq[l]);
    pack(Wk_[l], 356, 384, 128, 0, 128, BWk[l]);
    pack(Wv_[l], 356, 384, 128, 0, 128, BWv[l]);
    pack(Wo_[l], 256, 256, 128, 0, 128, BWo[l]);
  }

  // ---- GRU update of the M touched memory rows ----
  gru_kernel<<<Mc / 16, 32, 0, stream>>>(msgs, msg_nids, memory, BWih, BWhh,
                                         gru_bih, gru_bhh, upd);

  // ---- hop 1 (neighbor node feats = zeros) ----
  qproj_kernel<<<N1c / 16, 32, 0, stream>>>(memory, upd, map, seed1, time_b,
                                            BWq[1], bq1, qbuf, nf);
  attn_kernel<K1c><<<N1c / 2, 64, 0, stream>>>(qbuf, nullptr, nullptr,
                                               nbr_feats1, times1, nbr_times1,
                                               time_w, time_b, nbr_mask1,
                                               BWk[1], BWv[1], bk1, bv1, hout);
  oproj_kernel<<<N1c / 16, 32, 0, stream>>>(hout, nf, BWo[1], bo1, out1);

  // ---- hop 0 (neighbor node feats = out1 rows) ----
  qproj_kernel<<<N0c / 16, 32, 0, stream>>>(memory, upd, map, seed0, time_b,
                                            BWq[0], bq0, qbuf, nf);
  attn_kernel<K0c><<<N0c / 2, 64, 0, stream>>>(qbuf, out1, nbr_local0,
                                               nbr_feats0, times0, nbr_times0,
                                               time_w, time_b, nbr_mask0,
                                               BWk[0], BWv[0], bk0, bv0, hout);
  oproj_kernel<<<N0c / 16, 32, 0, stream>>>(hout, nf, BWo[0], bo0, out0);

  // ---- link prediction: pos then neg, concatenated in d_out ----
  float* outF = (float*)d_out;
  linkpred_kernel<<<Bc, 128, 0, stream>>>(out0, src_local, dst_local,
                                          lp_Ws, lp_bs, lp_Wd, lp_bd, lp_Wo, lp_bo,
                                          outF);
  linkpred_kernel<<<Bc, 128, 0, stream>>>(out0, src_local, neg_local,
                                          lp_Ws, lp_bs, lp_Wd, lp_bd, lp_Wo, lp_bo,
                                          outF + Bc);
}
